// TransformerBlockV2_83554293776783
// MI455X (gfx1250) — matrix operand order repaired and checked by reading, not compiled, not run on hardware
//
#include <hip/hip_runtime.h>
#include <hip/hip_bf16.h>
#include <math.h>

// ---------------- problem constants ----------------
#define WSZ   7
#define SS    3
#define HT    56
#define WT    56
#define NH    8
#define DIM   256
#define HD    32          // DIM / NH
#define SQ    49          // WSZ*WSZ
#define NWIN  64          // (HT/WSZ)*(WT/WSZ)
#define BATCH 32
#define NTOK  (BATCH * HT * WT)   // 100352 tokens
#define BW    (BATCH * NWIN)      // 2048 windows

typedef __attribute__((ext_vector_type(16))) __bf16 v16bf;
typedef __attribute__((ext_vector_type(8)))  float  v8f;

// ---------------- WMMA helpers (CDNA5 gfx1250, wave32) ----------------
__device__ __forceinline__ v8f wmma_bf16(v16bf a, v16bf b, v8f c) {
  // D = A(16x32 bf16) * B(32x16 bf16) + C(16x16 f32)
  return __builtin_amdgcn_wmma_f32_16x16x32_bf16(
      false, a, false, b, (short)0, c, false, false);
}

// A fragment, 16x32 bf16, row-major source with element stride `ld`.
// Layout: lanes 0-15 hold M=lane, K 0..15; lanes 16-31 hold
// M=lane-16, K 16..31 (matches load_b_frag).
__device__ __forceinline__ v16bf load_a_frag(const __bf16* base, int ld, int lane) {
  const int row = lane & 15;
  const int kb  = (lane >> 4) * 16;
  const __bf16* p = base + row * ld + kb;
  v16bf a;
#pragma unroll
  for (int i = 0; i < 8; ++i) a[i] = p[i];
#pragma unroll
  for (int i = 0; i < 8; ++i) a[8 + i] = p[8 + i];
  return a;
}

// B fragment, 32x16 bf16, from a K-contiguous (i.e. B^T row-major) buffer:
// element i of lane holds B[kb+i][col]; lanes 0-15: col=lane, K 0..15;
// lanes 16-31: col=lane-16, K 16..31.
__device__ __forceinline__ v16bf load_b_frag(const __bf16* baseT, int ld, int lane) {
  const int col = lane & 15;
  const int kb  = (lane >> 4) * 16;
  const __bf16* p = baseT + col * ld + kb;
  v16bf b;
#pragma unroll
  for (int i = 0; i < 16; ++i) b[i] = p[i];
  return b;
}

// ---------------- generic bf16-WMMA GEMM:  C = A @ W + bias ----------------
// A: M x K (TA = float or __bf16, row-major), W: K x N (f32 row-major).
// Block tile 128x64, BK=32, 256 threads = 8 waves (4x2 of 32x32 sub-tiles).
// EPI 0: f32 out (+bias).  EPI 1: exact GELU, bf16 out (+bias).
#define BM 128
#define BN 64
#define BK 32
#define LDA_S 34   // padded LDS strides (bf16 elems) to dodge bank conflicts
#define LDB_S 34

template <typename TA, int EPI>
__global__ __launch_bounds__(256) void gemm_wmma(
    const TA* __restrict__ A, const float* __restrict__ W,
    const float* __restrict__ bias, void* __restrict__ Cout,
    int M, int N, int K) {
  __shared__ __bf16 As[BM * LDA_S];
  __shared__ __bf16 BsT[BN * LDB_S];

  const int tid  = threadIdx.x;
  const int lane = tid & 31;
  const int w    = tid >> 5;
  const int wm   = w >> 1;     // 0..3
  const int wn   = w & 1;      // 0..1
  const int bm   = blockIdx.y * BM;
  const int bn   = blockIdx.x * BN;

  v8f c00 = {}, c01 = {}, c10 = {}, c11 = {};

  for (int kk = 0; kk < K; kk += BK) {
    // stage A tile (128x32): 16 contiguous elems per thread, cvt -> bf16
    {
      const int row = tid >> 1;
      const int k0  = (tid & 1) * 16;
      const TA* src = A + (size_t)(bm + row) * K + kk + k0;
      __bf16* dst = &As[row * LDA_S + k0];
#pragma unroll
      for (int i = 0; i < 16; ++i) dst[i] = (__bf16)(float)src[i];
    }
    // stage W tile (32x64) transposed into BsT[n][k]
    {
      const int kr = tid >> 3;           // 0..31
      const int n0 = (tid & 7) * 8;      // 0..56
      const float* src = W + (size_t)(kk + kr) * N + bn + n0;
#pragma unroll
      for (int i = 0; i < 8; ++i) BsT[(n0 + i) * LDB_S + kr] = (__bf16)src[i];
    }
    __syncthreads();

    v16bf a0 = load_a_frag(&As[(wm * 32 + 0)  * LDA_S], LDA_S, lane);
    v16bf a1 = load_a_frag(&As[(wm * 32 + 16) * LDA_S], LDA_S, lane);
    v16bf b0 = load_b_frag(&BsT[(wn * 32 + 0)  * LDB_S], LDB_S, lane);
    v16bf b1 = load_b_frag(&BsT[(wn * 32 + 16) * LDB_S], LDB_S, lane);
    c00 = wmma_bf16(a0, b0, c00);
    c01 = wmma_bf16(a0, b1, c01);
    c10 = wmma_bf16(a1, b0, c10);
    c11 = wmma_bf16(a1, b1, c11);
    __syncthreads();
  }

  // epilogue: C layout — lane<16: N=lane, M=v; lane>=16: N=lane-16, M=v+8
  const int colb = bn + wn * 32 + (lane & 15);
  const int rowb = bm + wm * 32 + ((lane >> 4) * 8);
#pragma unroll
  for (int t = 0; t < 4; ++t) {
    const v8f& c  = (t == 0) ? c00 : (t == 1) ? c01 : (t == 2) ? c10 : c11;
    const int col = colb + ((t & 1) * 16);
    const int r0  = rowb + ((t >> 1) * 16);
    const float bv = bias[col];
#pragma unroll
    for (int v = 0; v < 8; ++v) {
      const int row = r0 + v;
      float val = c[v] + bv;
      if (EPI == 1) {
        val = 0.5f * val * (1.0f + erff(val * 0.70710678118654752f));
        ((__bf16*)Cout)[(size_t)row * N + col] = (__bf16)val;
      } else {
        ((float*)Cout)[(size_t)row * N + col] = val;
      }
    }
  }
}

// ---------------- CPB bias table (Swin-V2 continuous position bias) --------
__global__ __launch_bounds__(256) void cpb_kernel(
    const float* __restrict__ w1, const float* __restrict__ b1,
    const float* __restrict__ w2, const float* __restrict__ b2,
    float* __restrict__ bias_out /* [NH][SQ][SQ] */) {
  __shared__ float tbl[169 * NH];
  const int tid = threadIdx.x;
  for (int r = tid; r < 169; r += 256) {
    const int i = r / 13, j = r % 13;
    const float g0 = (float)(i - 6) * (8.0f / 6.0f);
    const float g1 = (float)(j - 6) * (8.0f / 6.0f);
    const float s0 = (g0 > 0.f) ? 1.f : ((g0 < 0.f) ? -1.f : 0.f);
    const float s1 = (g1 > 0.f) ? 1.f : ((g1 < 0.f) ? -1.f : 0.f);
    const float t0 = s0 * __log2f(fabsf(g0) + 1.0f) * (1.0f / 3.0f);
    const float t1 = s1 * __log2f(fabsf(g1) + 1.0f) * (1.0f / 3.0f);
    float acc[NH];
#pragma unroll
    for (int h = 0; h < NH; ++h) acc[h] = b2[h];
    for (int jh = 0; jh < 512; ++jh) {
      const float hd = fmaxf(t0 * w1[jh] + t1 * w1[512 + jh] + b1[jh], 0.0f);
#pragma unroll
      for (int h = 0; h < NH; ++h) acc[h] += hd * w2[jh * NH + h];
    }
#pragma unroll
    for (int h = 0; h < NH; ++h) tbl[r * NH + h] = acc[h];
  }
  __syncthreads();
  for (int pq = tid; pq < SQ * SQ; pq += 256) {
    const int p = pq / SQ, q = pq % SQ;
    const int idx = (p / WSZ - q / WSZ + 6) * 13 + (p % WSZ - q % WSZ + 6);
#pragma unroll
    for (int h = 0; h < NH; ++h) {
      const float x = tbl[idx * NH + h];
      bias_out[h * (SQ * SQ) + pq] = 16.0f / (1.0f + __expf(-x));
    }
  }
}

// ---------------- windowed cosine attention: one wave per (window, head) ---
__global__ __launch_bounds__(32) void attn_kernel(
    const float* __restrict__ qkv,        // [NTOK][3*DIM], original token order
    const float* __restrict__ cpb_bias,   // [NH][SQ][SQ]
    const float* __restrict__ logit_scale,// [NH]
    float* __restrict__ attn_out)         // [NTOK][DIM], original token order
{
  const int head = blockIdx.x;            // 0..7
  const int win  = blockIdx.y;            // 0..2047
  const int lane = threadIdx.x;           // 0..31
  const int batch = win >> 6;
  const int nw    = win & 63;
  const int wi    = nw >> 3, wj = nw & 7;

  __shared__ __bf16 Qs[64 * LDA_S];       // normalized q, rows padded to 64
  __shared__ __bf16 Ks[64 * LDA_S];       // normalized k, row = key index
  __shared__ __bf16 VsT[32 * 66];         // VsT[d][key]
  __shared__ float  Smat[64 * 65];
  __shared__ __bf16 Ps[64 * 66];

  auto tok = [&](int p) {                 // window token -> original token id
    const int a = p / WSZ, b = p - a * WSZ;
    int h0 = wi * WSZ + a + SS; if (h0 >= HT) h0 -= HT;   // undo roll(-SS)
    int w0 = wj * WSZ + b + SS; if (w0 >= WT) w0 -= WT;
    return batch * (HT * WT) + h0 * WT + w0;
  };
  auto clsof = [&](int p) {               // shifted-window mask class
    const int a = p / WSZ, b = p - a * WSZ;
    const int hh = wi * WSZ + a, ww = wj * WSZ + b;
    const int ch = hh < (HT - WSZ) ? 0 : (hh < (HT - SS) ? 1 : 2);
    const int cw = ww < (WT - WSZ) ? 0 : (ww < (WT - SS) ? 1 : 2);
    return ch * 3 + cw;
  };

  // gather + cosine-normalize q,k; gather v (transposed)
  for (int r = 0; r < SQ; ++r) {
    const size_t base = (size_t)tok(r) * (3 * DIM) + head * HD + lane;
    const float q  = qkv[base];
    const float k  = qkv[base + DIM];
    const float vv = qkv[base + 2 * DIM];
    float qs = q * q, ks = k * k;
#pragma unroll
    for (int off = 16; off; off >>= 1) {
      qs += __shfl_xor(qs, off, 32);
      ks += __shfl_xor(ks, off, 32);
    }
    Qs[r * LDA_S + lane]  = (__bf16)(q / fmaxf(sqrtf(qs), 1e-12f));
    Ks[r * LDA_S + lane]  = (__bf16)(k / fmaxf(sqrtf(ks), 1e-12f));
    VsT[lane * 66 + r]    = (__bf16)vv;
  }
  for (int r = SQ; r < 64; ++r) {         // zero pad
    Qs[r * LDA_S + lane] = (__bf16)0.0f;
    Ks[r * LDA_S + lane] = (__bf16)0.0f;
    VsT[lane * 66 + r]   = (__bf16)0.0f;
  }
  __syncthreads();

  // S = Qn @ Kn^T  (64x64 padded, K=32) — 16 WMMA
#pragma unroll
  for (int mi = 0; mi < 4; ++mi) {
#pragma unroll
    for (int ni = 0; ni < 4; ++ni) {
      v16bf a = load_a_frag(&Qs[mi * 16 * LDA_S], LDA_S, lane);
      v16bf b = load_b_frag(&Ks[ni * 16 * LDA_S], LDA_S, lane);
      v8f c = {};
      c = wmma_bf16(a, b, c);
      const int col = ni * 16 + (lane & 15);
      const int r0  = mi * 16 + ((lane >> 4) * 8);
#pragma unroll
      for (int v = 0; v < 8; ++v) Smat[(r0 + v) * 65 + col] = c[v];
    }
  }
  __syncthreads();

  const float scale = __expf(fminf(logit_scale[head], 4.6051701859880914f));
  const float* bh = cpb_bias + head * (SQ * SQ);

  // fused scale + CPB bias + shift mask + softmax, emit bf16 P
  for (int r = 0; r < SQ; ++r) {
    const int rcls = clsof(r);
    const int q0 = lane, q1 = lane + 32;
    float v0 = Smat[r * 65 + q0] * scale + bh[r * SQ + q0] +
               ((clsof(q0) == rcls) ? 0.0f : -100.0f);
    float v1 = -3.0e38f;
    if (q1 < SQ)
      v1 = Smat[r * 65 + q1] * scale + bh[r * SQ + q1] +
           ((clsof(q1) == rcls) ? 0.0f : -100.0f);
    float m = fmaxf(v0, v1);
#pragma unroll
    for (int off = 16; off; off >>= 1) m = fmaxf(m, __shfl_xor(m, off, 32));
    const float e0 = __expf(v0 - m);
    const float e1 = (q1 < SQ) ? __expf(v1 - m) : 0.0f;
    float s = e0 + e1;
#pragma unroll
    for (int off = 16; off; off >>= 1) s += __shfl_xor(s, off, 32);
    const float inv = 1.0f / s;
    Ps[r * 66 + q0] = (__bf16)(e0 * inv);
    Ps[r * 66 + q1] = (__bf16)(e1 * inv);   // zero for q1 >= SQ
  }
  for (int r = SQ; r < 64; ++r) {           // zero pad rows
    Ps[r * 66 + lane]      = (__bf16)0.0f;
    Ps[r * 66 + lane + 32] = (__bf16)0.0f;
  }
  __syncthreads();

  // O = P @ V  (64x32 padded, K=64) — 16 WMMA; scatter rows < SQ
#pragma unroll
  for (int mi = 0; mi < 4; ++mi) {
#pragma unroll
    for (int ni = 0; ni < 2; ++ni) {
      v8f c = {};
#pragma unroll
      for (int kk = 0; kk < 64; kk += 32) {
        v16bf a = load_a_frag(&Ps[mi * 16 * 66 + kk], 66, lane);
        v16bf b = load_b_frag(&VsT[ni * 16 * 66 + kk], 66, lane);
        c = wmma_bf16(a, b, c);
      }
      const int col = ni * 16 + (lane & 15);
      const int r0  = mi * 16 + ((lane >> 4) * 8);
#pragma unroll
      for (int v = 0; v < 8; ++v) {
        const int row = r0 + v;
        if (row < SQ)
          attn_out[(size_t)tok(row) * DIM + head * HD + col] = c[v];
      }
    }
  }
}

// ---------------- LayerNorm + residual (one wave per token) ----------------
__global__ __launch_bounds__(256) void ln_residual(
    const float* __restrict__ in, const float* __restrict__ res,
    const float* __restrict__ g, const float* __restrict__ b,
    float* __restrict__ out, int M) {
  const int wave = threadIdx.x >> 5;
  const int lane = threadIdx.x & 31;
  const int t = blockIdx.x * 8 + wave;
  if (t >= M) return;
  const float* x = in + (size_t)t * DIM + lane * 8;
  float v[8];
  float s = 0.f, s2 = 0.f;
#pragma unroll
  for (int i = 0; i < 8; ++i) { v[i] = x[i]; s += v[i]; s2 += v[i] * v[i]; }
#pragma unroll
  for (int off = 16; off; off >>= 1) {
    s  += __shfl_xor(s, off, 32);
    s2 += __shfl_xor(s2, off, 32);
  }
  const float mu  = s * (1.0f / DIM);
  const float var = s2 * (1.0f / DIM) - mu * mu;
  const float inv = rsqrtf(var + 1e-5f);
  const int base = lane * 8;
  float* o = out + (size_t)t * DIM + base;
  const float* rr = res + (size_t)t * DIM + base;
#pragma unroll
  for (int i = 0; i < 8; ++i)
    o[i] = (v[i] - mu) * inv * g[base + i] + b[base + i] + rr[i];
}

// ---------------- host launcher ----------------
extern "C" void kernel_launch(void* const* d_in, const int* in_sizes, int n_in,
                              void* d_out, int out_size, void* d_ws, size_t ws_size,
                              hipStream_t stream) {
  const float* x           = (const float*)d_in[0];
  const float* qkv_w       = (const float*)d_in[1];
  const float* qkv_b       = (const float*)d_in[2];
  const float* proj_w      = (const float*)d_in[3];
  const float* proj_b      = (const float*)d_in[4];
  const float* logit_scale = (const float*)d_in[5];
  const float* cpb_w1      = (const float*)d_in[6];
  const float* cpb_b1      = (const float*)d_in[7];
  const float* cpb_w2      = (const float*)d_in[8];
  const float* cpb_b2      = (const float*)d_in[9];
  const float* ln1_g       = (const float*)d_in[10];
  const float* ln1_b       = (const float*)d_in[11];
  const float* mlp_w1      = (const float*)d_in[12];
  const float* mlp_b1      = (const float*)d_in[13];
  const float* mlp_w2      = (const float*)d_in[14];
  const float* mlp_b2      = (const float*)d_in[15];
  const float* ln2_g       = (const float*)d_in[16];
  const float* ln2_b       = (const float*)d_in[17];

  const int M = NTOK;   // 100352, divisible by BM=128
  char* ws = (char*)d_ws;
  size_t off = 0;
  auto carve = [&](size_t bytes) {
    void* p = ws + off;
    off += (bytes + 255) & ~(size_t)255;
    return p;
  };
  float*  cpbias   = (float*)carve((size_t)NH * SQ * SQ * 4);
  float*  qkv_buf  = (float*)carve((size_t)M * 3 * DIM * 4);
  float*  attn_out = (float*)carve((size_t)M * DIM * 4);
  float*  proj_out = (float*)carve((size_t)M * DIM * 4);
  float*  xo       = (float*)carve((size_t)M * DIM * 4);
  __bf16* h1       = (__bf16*)carve((size_t)M * 4 * DIM * 2);
  float*  h2       = attn_out;   // reuse (attn_out dead after proj GEMM)

  cpb_kernel<<<1, 256, 0, stream>>>(cpb_w1, cpb_b1, cpb_w2, cpb_b2, cpbias);

  // QKV:  [M,256] @ [256,768]
  gemm_wmma<float, 0><<<dim3(3 * DIM / BN, M / BM), 256, 0, stream>>>(
      x, qkv_w, qkv_b, qkv_buf, M, 3 * DIM, DIM);

  // windowed attention
  attn_kernel<<<dim3(NH, BW), 32, 0, stream>>>(qkv_buf, cpbias, logit_scale,
                                               attn_out);

  // proj: [M,256] @ [256,256]
  gemm_wmma<float, 0><<<dim3(DIM / BN, M / BM), 256, 0, stream>>>(
      attn_out, proj_w, proj_b, proj_out, M, DIM, DIM);

  // xo = LN1(proj_out) + x
  ln_residual<<<(M + 7) / 8, 256, 0, stream>>>(proj_out, x, ln1_g, ln1_b, xo, M);

  // h1 = gelu(xo @ W1 + b1)  -> bf16
  gemm_wmma<float, 1><<<dim3(4 * DIM / BN, M / BM), 256, 0, stream>>>(
      xo, mlp_w1, mlp_b1, h1, M, 4 * DIM, DIM);

  // h2 = h1 @ W2 + b2
  gemm_wmma<__bf16, 0><<<dim3(DIM / BN, M / BM), 256, 0, stream>>>(
      h1, mlp_w2, mlp_b2, h2, M, DIM, 4 * DIM);

  // out = LN2(h2) + xo
  ln_residual<<<(M + 7) / 8, 256, 0, stream>>>(h2, xo, ln2_g, ln2_b,
                                               (float*)d_out, M);
}
